// CTCalone_63419487093326
// MI455X (gfx1250) — matrix operand order repaired and checked by reading, not compiled, not run on hardware
//
#include <hip/hip_runtime.h>

// ---------------------------------------------------------------------------
// CDNA5 / gfx1250 implementation of the attention + BiLSTM + softmax pipeline
// B=32, T=512, D=512, H=256, V=96.
//  - all heavy matmuls: v_wmma_f32_16x16x32_bf16 (f32 accumulate)
//  - NT layout everywhere (second operand pre-transposed) so fragment loads
//    are contiguous 16B chunks per lane
//  - GEMM stages tiles in double-buffered LDS via async global->LDS DMA
//    (GLOBAL_LOAD_ASYNC_TO_LDS_B128, ASYNCcnt / s_wait_asynccnt)
// ---------------------------------------------------------------------------

typedef __bf16  v16bf __attribute__((ext_vector_type(16)));
typedef float   v8f   __attribute__((ext_vector_type(8)));
typedef unsigned int u32x4 __attribute__((ext_vector_type(4)));
typedef int     v4i   __attribute__((ext_vector_type(4)));

union Frag {
    v16bf v;
    u32x4 q[2];
};

__device__ __forceinline__ unsigned short f2bf(float f) {
    unsigned u = __float_as_uint(f);
    unsigned r = u + 0x7FFFu + ((u >> 16) & 1u);   // round-to-nearest-even
    return (unsigned short)(r >> 16);
}
__device__ __forceinline__ float bf2f(unsigned short h) {
    return __uint_as_float(((unsigned)h) << 16);
}
__device__ __forceinline__ float sigm(float x) {
    return 1.0f / (1.0f + __expf(-x));
}
__device__ __forceinline__ v8f wmma_bf16(v16bf a, v16bf b, v8f c) {
    return __builtin_amdgcn_wmma_f32_16x16x32_bf16(
        false, a, false, b, (short)0, c, false, false);
}

// ---------------------------------------------------------------------------
// Async global->LDS 16B copy (CDNA5 GLOBAL_LOAD_ASYNC_TO_LDS_B128), gated so
// the file always compiles: fall back to a plain load/store copy otherwise.
// Builtin signature (from hipcc diagnostic): param0 = v4i in AS1 (global),
// so: (v4i AS1* src, v4i AS3* dstLDS, imm offset, imm cpol).
// ---------------------------------------------------------------------------
#if defined(__has_builtin)
#if __has_builtin(__builtin_amdgcn_global_load_async_to_lds_b128)
#define HAVE_ASYNC_LDS 1
#endif
#if __has_builtin(__builtin_amdgcn_s_wait_asynccnt)
#define HAVE_WAIT_ASYNC 1
#endif
#endif

#if defined(HAVE_ASYNC_LDS)
typedef __attribute__((address_space(1))) v4i* g_v4i_ptr;
typedef __attribute__((address_space(3))) v4i* l_v4i_ptr;
#endif

__device__ __forceinline__ void cp16_to_lds(unsigned short* dst,
                                            const unsigned short* src) {
#if defined(HAVE_ASYNC_LDS)
    // generic LDS pointer: low 32 bits are the LDS byte offset
    __builtin_amdgcn_global_load_async_to_lds_b128(
        (g_v4i_ptr)(size_t)src, (l_v4i_ptr)(unsigned)(size_t)dst, 0, 0);
#else
    *(u32x4*)dst = *(const u32x4*)src;
#endif
}

__device__ __forceinline__ void cp_wait_all() {
#if defined(HAVE_ASYNC_LDS)
#if defined(HAVE_WAIT_ASYNC)
    __builtin_amdgcn_s_wait_asynccnt(0);
#else
    asm volatile("s_wait_asynccnt 0x0" ::: "memory");
#endif
#endif
}

// ---------------------------------------------------------------------------
// Generic batched NT GEMM: C[m,n] = sum_k A[m,k]*B[n,k] (A,B bf16, C f32
// accumulate, stored bf16 or f32).  Block = 256 thr = 8 waves (2x4), block
// tile 64 x 256, wave tile 32 x 64 (2x4 WMMA tiles), K-step 32, tiles staged
// in double-buffered LDS via async DMA.  Requires M%64==0, K%32==0, N%16==0.
// Out-of-range B rows are clamped during copy (always in-bounds); compute is
// fully branch-free; stores masked by a uniform per-tile test.
// ---------------------------------------------------------------------------
__global__ __launch_bounds__(256) void gemm_nt_kernel(
    const unsigned short* __restrict__ A, const unsigned short* __restrict__ B,
    void* __restrict__ Cout, int M, int N, int K, int lda, int ldb, int ldc,
    long long sA, long long sB, long long sC, int out_bf16)
{
    __shared__ __align__(16) unsigned short A_s[2][64 * 32];    //  8 KB
    __shared__ __align__(16) unsigned short B_s[2][256 * 32];   // 32 KB

    const int batch = blockIdx.z;
    A += (long long)batch * sA;
    B += (long long)batch * sB;
    const long long cb = (long long)batch * sC;

    const int tid  = threadIdx.x;
    const int wave = tid >> 5;
    const int lane = tid & 31;
    const int wm = wave >> 2, wn = wave & 3;
    const int mblk = blockIdx.x * 64;
    const int nblk = blockIdx.y * 256;
    const int half = lane >> 4;
    const int l16  = lane & 15;
    const int Nc   = N - 1;

    // tile copy: A tile = 256 x 16B chunks (one per thread), B tile = 1024
    auto copy_tiles = [&](int ph, int k0) {
        {
            int r = tid >> 2, s = (tid & 3) * 8;
            cp16_to_lds(&A_s[ph][r * 32 + s],
                        A + (size_t)(mblk + r) * lda + k0 + s);
        }
        #pragma unroll
        for (int i = 0; i < 4; ++i) {
            int cc = tid + i * 256;
            int r = cc >> 2, s = (cc & 3) * 8;
            int rb = nblk + r;
            if (rb > Nc) rb = Nc;                 // clamp: stay in-bounds
            cp16_to_lds(&B_s[ph][r * 32 + s],
                        B + (size_t)rb * ldb + k0 + s);
        }
    };

    v8f acc[2][4];
    #pragma unroll
    for (int m = 0; m < 2; ++m)
        #pragma unroll
        for (int i = 0; i < 4; ++i)
            acc[m][i] = (v8f){0.f, 0.f, 0.f, 0.f, 0.f, 0.f, 0.f, 0.f};

    const int KT = K >> 5;
    copy_tiles(0, 0);

    for (int kt = 0; kt < KT; ++kt) {
        cp_wait_all();            // own wave's pending copies complete
        __syncthreads();          // everyone's copies complete / buffer free
        if (kt + 1 < KT)
            copy_tiles((kt + 1) & 1, (kt + 1) << 5);   // overlaps compute

        const unsigned short* As = A_s[kt & 1];
        const unsigned short* Bs = B_s[kt & 1];

        Frag a0, a1, bf[4];
        {
            const unsigned short* p = As + (wm * 32 + l16) * 32 + half * 8;
            a0.q[0] = *(const u32x4*)(p);
            a0.q[1] = *(const u32x4*)(p + 16);
            const unsigned short* q = As + (wm * 32 + 16 + l16) * 32 + half * 8;
            a1.q[0] = *(const u32x4*)(q);
            a1.q[1] = *(const u32x4*)(q + 16);
        }
        #pragma unroll
        for (int i = 0; i < 4; ++i) {
            const unsigned short* p = Bs + (wn * 64 + i * 16 + l16) * 32 + half * 8;
            bf[i].q[0] = *(const u32x4*)(p);
            bf[i].q[1] = *(const u32x4*)(p + 16);
        }
        #pragma unroll
        for (int i = 0; i < 4; ++i) {
            acc[0][i] = wmma_bf16(a0.v, bf[i].v, acc[0][i]);
            acc[1][i] = wmma_bf16(a1.v, bf[i].v, acc[1][i]);
        }
    }

    // C/D layout: VGPR v of lane L holds C[mtile + v + 8*half][ntile + (L&15)]
    #pragma unroll
    for (int m = 0; m < 2; ++m) {
        #pragma unroll
        for (int i = 0; i < 4; ++i) {
            int ntile = nblk + wn * 64 + i * 16;
            if (ntile >= N) continue;             // uniform per tile (N%16==0)
            int col = ntile + l16;
            #pragma unroll
            for (int v = 0; v < 8; ++v) {
                int row = mblk + wm * 32 + m * 16 + v + half * 8;
                float val = acc[m][i][v];
                if (out_bf16)
                    ((unsigned short*)Cout)[cb + (long long)row * ldc + col] = f2bf(val);
                else
                    ((float*)Cout)[cb + (long long)row * ldc + col] = val;
            }
        }
    }
}

// ---------------------------------------------------------------------------
// Row-normalize X along D, emit Xn (bf16) and X^T (bf16, per batch D x T).
// One wave per (b,t) row.
// ---------------------------------------------------------------------------
__global__ __launch_bounds__(256) void normalize_transpose_kernel(
    const float* __restrict__ X, unsigned short* __restrict__ Xn,
    unsigned short* __restrict__ Xt)
{
    int wave = (int)((blockIdx.x * blockDim.x + threadIdx.x) >> 5);
    int lane = threadIdx.x & 31;
    if (wave >= 32 * 512) return;
    int b = wave >> 9, t = wave & 511;
    const float* row = X + (size_t)wave * 512;

    float x[16];
    float s = 0.f;
    #pragma unroll
    for (int e = 0; e < 16; ++e) {
        x[e] = row[e * 32 + lane];
        s += x[e] * x[e];
    }
    #pragma unroll
    for (int o = 16; o > 0; o >>= 1) s += __shfl_xor(s, o, 32);
    float inv = 1.0f / (sqrtf(s) + 1e-12f);

    #pragma unroll
    for (int e = 0; e < 16; ++e) {
        int d = e * 32 + lane;
        Xn[(size_t)wave * 512 + d] = f2bf(x[e] * inv);
        Xt[((size_t)b * 512 + d) * 512 + t] = f2bf(x[e]);
    }
}

// ---------------------------------------------------------------------------
// ADJ layer: A_D[s,u] = exp(-|s-u|/8)/rowsum(s); store A_D^T in bf16.
// ---------------------------------------------------------------------------
__global__ void adj_rowsum_kernel(float* __restrict__ rowsum)
{
    int s = blockIdx.x * blockDim.x + threadIdx.x;
    if (s >= 512) return;
    float acc = 0.f;
    for (int u = 0; u < 512; ++u)
        acc += __expf(-fabsf((float)(s - u)) * 0.125f);
    rowsum[s] = acc;
}

__global__ void adj_fill_kernel(const float* __restrict__ rowsum,
                                unsigned short* __restrict__ ADt)
{
    int i = blockIdx.x * blockDim.x + threadIdx.x;   // i = u*512 + s
    if (i >= 512 * 512) return;
    int u = i >> 9, s = i & 511;
    float v = __expf(-fabsf((float)(s - u)) * 0.125f) / rowsum[s];
    ADt[i] = f2bf(v);
}

// ---------------------------------------------------------------------------
// Transpose f32 [R x C] -> bf16 [Cpad x R] (zero pad rows >= C).
// ---------------------------------------------------------------------------
__global__ void transpose_f32_to_bf16(const float* __restrict__ src,
                                      unsigned short* __restrict__ dst,
                                      int R, int C, int Cpad)
{
    int i = blockIdx.x * blockDim.x + threadIdx.x;   // i = c*R + r
    if (i >= R * Cpad) return;
    int c = i / R, r = i % R;
    dst[i] = (c < C) ? f2bf(src[(size_t)r * C + c]) : (unsigned short)0;
}

// ---------------------------------------------------------------------------
// Persistent BiLSTM kernel: grid = 2 (direction), block = 256 (8 waves).
// h (bf16) lives in LDS and feeds WMMA A-fragments; z (f32) staged in LDS;
// cell state c lives in registers (thread j owns hidden column j, 32 batches).
// ---------------------------------------------------------------------------
__global__ __launch_bounds__(256) void lstm_kernel(
    const unsigned short* __restrict__ Zx_f, const unsigned short* __restrict__ Zx_b,
    const unsigned short* __restrict__ WhT_f, const unsigned short* __restrict__ WhT_b,
    const float* __restrict__ bias_f, const float* __restrict__ bias_b,
    unsigned short* __restrict__ hcat)
{
    __shared__ __align__(16) float          z_s[32 * 1024];   // 128 KB
    __shared__ __align__(16) unsigned short h_s[32 * 256];    //  16 KB

    const int dir = blockIdx.x;
    const unsigned short* Zx   = dir ? Zx_b  : Zx_f;
    const unsigned short* WhT  = dir ? WhT_b : WhT_f;
    const float*          bias = dir ? bias_b : bias_f;

    const int tid  = threadIdx.x;
    const int lane = tid & 31;
    const int wave = tid >> 5;
    const int half = lane >> 4;
    const int l16  = lane & 15;

    for (int i = tid; i < 32 * 256; i += 256) h_s[i] = 0;
    float c_reg[32];
    #pragma unroll
    for (int i = 0; i < 32; ++i) c_reg[i] = 0.f;
    __syncthreads();

    for (int step = 0; step < 512; ++step) {
        const int t = dir ? (511 - step) : step;

        // ---- Phase A: z = h @ Wh + Zx[t] + bias (WMMA over K=256) ----
        for (int nt = 0; nt < 8; ++nt) {
            const int ncol = wave * 128 + nt * 16 + l16;
            v8f acc0 = (v8f){0.f, 0.f, 0.f, 0.f, 0.f, 0.f, 0.f, 0.f};
            v8f acc1 = acc0;
            #pragma unroll
            for (int k0 = 0; k0 < 256; k0 += 32) {
                Frag fa0, fa1, fb;
                const unsigned short* pa0 = h_s + l16 * 256 + k0 + half * 8;
                fa0.q[0] = *(const u32x4*)(pa0);
                fa0.q[1] = *(const u32x4*)(pa0 + 16);
                const unsigned short* pa1 = h_s + (16 + l16) * 256 + k0 + half * 8;
                fa1.q[0] = *(const u32x4*)(pa1);
                fa1.q[1] = *(const u32x4*)(pa1 + 16);
                const unsigned short* pb = WhT + (size_t)ncol * 256 + k0 + half * 8;
                fb.q[0] = *(const u32x4*)(pb);
                fb.q[1] = *(const u32x4*)(pb + 16);
                acc0 = wmma_bf16(fa0.v, fb.v, acc0);
                acc1 = wmma_bf16(fa1.v, fb.v, acc1);
            }
            const float bval = bias[ncol];
            #pragma unroll
            for (int v = 0; v < 8; ++v) {
                int r0 = v + half * 8;          // batch rows 0..15
                int r1 = 16 + r0;               // batch rows 16..31
                float zx0 = bf2f(Zx[((size_t)r0 * 512 + t) * 1024 + ncol]);
                float zx1 = bf2f(Zx[((size_t)r1 * 512 + t) * 1024 + ncol]);
                z_s[r0 * 1024 + ncol] = acc0[v] + zx0 + bval;
                z_s[r1 * 1024 + ncol] = acc1[v] + zx1 + bval;
            }
        }
        __syncthreads();

        // ---- Phase B: gate math; thread j owns hidden column j ----
        const int j = tid;
        #pragma unroll 4
        for (int bb = 0; bb < 32; ++bb) {
            float zi = z_s[bb * 1024 + j];
            float zf = z_s[bb * 1024 + 256 + j];
            float zg = z_s[bb * 1024 + 512 + j];
            float zo = z_s[bb * 1024 + 768 + j];
            float cc = sigm(zf) * c_reg[bb] + sigm(zi) * tanhf(zg);
            c_reg[bb] = cc;
            float hh = sigm(zo) * tanhf(cc);
            unsigned short hb = f2bf(hh);
            h_s[bb * 256 + j] = hb;
            hcat[((size_t)bb * 512 + t) * 512 + dir * 256 + j] = hb;
        }
        __syncthreads();
    }
}

// ---------------------------------------------------------------------------
// Softmax over 97 classes (+bc bias), wave per (b,t) row; reads padded logits
// (ld=112), writes final f32 output [32,512,97].
// ---------------------------------------------------------------------------
__global__ __launch_bounds__(256) void softmax_kernel(
    const float* __restrict__ logits, const float* __restrict__ bc,
    float* __restrict__ out)
{
    int wave = (int)((blockIdx.x * blockDim.x + threadIdx.x) >> 5);
    int lane = threadIdx.x & 31;
    if (wave >= 32 * 512) return;
    const float* lrow = logits + (size_t)wave * 112;

    float v[4];
    float m = -3.4e38f;
    #pragma unroll
    for (int k = 0; k < 4; ++k) {
        int c = k * 32 + lane;
        v[k] = (c < 97) ? (lrow[c] + bc[c]) : -3.4e38f;
        m = fmaxf(m, v[k]);
    }
    #pragma unroll
    for (int o = 16; o > 0; o >>= 1) m = fmaxf(m, __shfl_xor(m, o, 32));
    float s = 0.f;
    #pragma unroll
    for (int k = 0; k < 4; ++k) {
        int c = k * 32 + lane;
        v[k] = (c < 97) ? __expf(v[k] - m) : 0.f;
        s += v[k];
    }
    #pragma unroll
    for (int o = 16; o > 0; o >>= 1) s += __shfl_xor(s, o, 32);
    float inv = 1.0f / s;
    #pragma unroll
    for (int k = 0; k < 4; ++k) {
        int c = k * 32 + lane;
        if (c < 97) out[(size_t)wave * 97 + c] = v[k] * inv;
    }
}

// ---------------------------------------------------------------------------
// Host-side orchestration
// ---------------------------------------------------------------------------
extern "C" void kernel_launch(void* const* d_in, const int* in_sizes, int n_in,
                              void* d_out, int out_size, void* d_ws, size_t ws_size,
                              hipStream_t stream)
{
    (void)in_sizes; (void)n_in; (void)out_size; (void)ws_size;

    const float* X    = (const float*)d_in[0];
    const float* Wx_f = (const float*)d_in[1];
    const float* Wh_f = (const float*)d_in[2];
    const float* b_f  = (const float*)d_in[3];
    const float* Wx_b = (const float*)d_in[4];
    const float* Wh_b = (const float*)d_in[5];
    const float* b_b  = (const float*)d_in[6];
    const float* Wc   = (const float*)d_in[7];
    const float* bc   = (const float*)d_in[8];
    // d_in[9..11] (label / lengths) feed only the CTC side branch -> unused.

    char* ws = (char*)d_ws;
    size_t off = 0;
    auto wsa = [&](size_t bytes) -> char* {
        char* p = ws + off;
        off += (bytes + 255) & ~(size_t)255;
        return p;
    };

    unsigned short* Xn    = (unsigned short*)wsa(32ull * 512 * 512 * 2);  // [B,T,D] bf16
    unsigned short* Xt    = (unsigned short*)wsa(32ull * 512 * 512 * 2);  // [B,D,T] bf16
    unsigned short* ADt   = (unsigned short*)wsa(512ull * 512 * 2);       // A_D^T bf16
    unsigned short* AS    = (unsigned short*)wsa(32ull * 512 * 512 * 2);  // A_S bf16
    unsigned short* Amat  = (unsigned short*)wsa(32ull * 512 * 512 * 2);  // A   bf16
    unsigned short* Xa    = (unsigned short*)wsa(32ull * 512 * 512 * 2);  // Xa  bf16
    unsigned short* WxTf  = (unsigned short*)wsa(1024ull * 512 * 2);      // Wx^T f
    unsigned short* WxTb  = (unsigned short*)wsa(1024ull * 512 * 2);      // Wx^T b
    unsigned short* WhTf  = (unsigned short*)wsa(1024ull * 256 * 2);      // Wh^T f
    unsigned short* WhTb  = (unsigned short*)wsa(1024ull * 256 * 2);      // Wh^T b
    unsigned short* WcT   = (unsigned short*)wsa(112ull * 512 * 2);       // Wc^T (pad 112)
    unsigned short* Zxf   = (unsigned short*)wsa(16384ull * 1024 * 2);    // Xa@Wx_f bf16
    unsigned short* Zxb   = (unsigned short*)wsa(16384ull * 1024 * 2);    // Xa@Wx_b bf16
    unsigned short* hcat  = (unsigned short*)wsa(16384ull * 512 * 2);     // [b,t,2H] bf16
    float*          logit = (float*)wsa(16384ull * 112 * 4);              // padded logits
    float*          rsum  = (float*)wsa(512ull * 4);

    // 1. ADJ matrix (A_D^T)
    adj_rowsum_kernel<<<2, 256, 0, stream>>>(rsum);
    adj_fill_kernel<<<(512 * 512) / 256, 256, 0, stream>>>(rsum, ADt);

    // 2. normalize + transpose X
    normalize_transpose_kernel<<<2048, 256, 0, stream>>>(X, Xn, Xt);

    // 3. weight transposes (f32 -> bf16)
    transpose_f32_to_bf16<<<(512 * 1024) / 256, 256, 0, stream>>>(Wx_f, WxTf, 512, 1024, 1024);
    transpose_f32_to_bf16<<<(512 * 1024) / 256, 256, 0, stream>>>(Wx_b, WxTb, 512, 1024, 1024);
    transpose_f32_to_bf16<<<(256 * 1024) / 256, 256, 0, stream>>>(Wh_f, WhTf, 256, 1024, 1024);
    transpose_f32_to_bf16<<<(256 * 1024) / 256, 256, 0, stream>>>(Wh_b, WhTb, 256, 1024, 1024);
    transpose_f32_to_bf16<<<(512 * 112) / 256, 256, 0, stream>>>(Wc, WcT, 512, 97, 112);

    auto gemm = [&](const unsigned short* A, const unsigned short* B, void* C,
                    int M, int N, int K, int lda, int ldb, int ldc,
                    long long sA, long long sB, long long sC, int obf, int batch) {
        dim3 grid((M + 63) / 64, (N + 255) / 256, batch);
        gemm_nt_kernel<<<grid, 256, 0, stream>>>(A, B, C, M, N, K, lda, ldb, ldc,
                                                 sA, sB, sC, obf);
    };

    const long long SQ = 512ll * 512;

    // 4. A_S = Xn @ Xn^T          (batched 512x512x512)
    gemm(Xn, Xn, AS, 512, 512, 512, 512, 512, 512, SQ, SQ, SQ, 1, 32);
    // 5. A = A_S @ A_D            (B = A_D^T, shared across batch)
    gemm(AS, ADt, Amat, 512, 512, 512, 512, 512, 512, SQ, 0, SQ, 1, 32);
    // 6. Xa = A @ X               (B = X^T per batch)
    gemm(Amat, Xt, Xa, 512, 512, 512, 512, 512, 512, SQ, SQ, SQ, 1, 32);
    // 7. gate precompute Zx = Xa2d @ Wx (both directions)
    gemm(Xa, WxTf, Zxf, 16384, 1024, 512, 512, 512, 1024, 0, 0, 0, 1, 1);
    gemm(Xa, WxTb, Zxb, 16384, 1024, 512, 512, 512, 1024, 0, 0, 0, 1, 1);

    // 8. BiLSTM recurrence (persistent, 1 block per direction)
    lstm_kernel<<<2, 256, 0, stream>>>(Zxf, Zxb, WhTf, WhTb, b_f, b_b, hcat);

    // 9. classifier logits = hcat @ Wc (N padded to 112)
    gemm(hcat, WcT, logit, 16384, 112, 512, 512, 512, 112, 0, 0, 0, 0, 1);

    // 10. softmax(+bc) -> d_out [32,512,97] f32
    softmax_kernel<<<2048, 256, 0, stream>>>(logit, bc, (float*)d_out);
}
